// AVENet_ssltie_27865747817250
// MI455X (gfx1250) — matrix operand order repaired and checked by reading, not compiled, not run on hardware
//
#include <hip/hip_runtime.h>
#include <hip/hip_bf16.h>
#include <stdint.h>

typedef __attribute__((ext_vector_type(16))) _Float16 v16h;
typedef __attribute__((ext_vector_type(8)))  _Float16 v8h;
typedef __attribute__((ext_vector_type(8)))  float    v8f;

#define TAU_     0.03f
#define EPS_     0.65f
#define EPS2_    0.40f
#define LOGIT_T_ 0.07f

#define NB   256     // batch
#define NC   512     // channels
#define HW   196     // 14*14
#define HWP  208     // padded to 13*16
#define NT   13      // M-tiles per wave
#define RS   40      // LDS row stride in f16 (80B: 16B-aligned rows, conflict-free)

// output flat offsets (A, logits, Pos, Neg, A0_ref)
#define OFF_A    0
#define OFF_LOG  50176
#define OFF_POS  116224
#define OFF_NEG  166400
#define OFF_REF  216576

__device__ __forceinline__ float sigf(float x) {
  return 1.0f / (1.0f + __expf(-x));
}

__device__ __forceinline__ v16h cat16(v8h lo, v8h hi) {
  return __builtin_shufflevector(lo, hi,
         0,1,2,3,4,5,6,7,8,9,10,11,12,13,14,15);
}

// async-copy one 208x32 K-slice (valid rows only) from imgT16 into LDS buffer
__device__ __forceinline__ void stage_async(_Float16* dst,
                                            const _Float16* src_kb,
                                            int tid) {
  // 196 rows x 64B, 16B chunks -> 784 chunks over 128 threads
  for (int i = tid; i < HW * 4; i += 128) {
    int hw = i >> 2, j = i & 3;
    unsigned lds = (unsigned)(unsigned long long)
        (const void*)(dst + hw * RS + j * 8);
    unsigned long long ga = (unsigned long long)
        (const void*)(src_kb + (size_t)hw * NC + j * 8);
    asm volatile("global_load_async_to_lds_b128 %0, %1, off"
                 :: "v"(lds), "v"(ga) : "memory");
  }
}

__device__ __forceinline__ void wait_async0() {
  asm volatile("s_wait_asynccnt 0x0" ::: "memory");
}

// --- kernel 1: L2-normalize audio rows -> f16 [256,512] -------------------
__global__ void norm_aud_kernel(const float* __restrict__ aud,
                                _Float16* __restrict__ aud16) {
  const int b = blockIdx.x;
  const int lane = threadIdx.x;
  const float* row = aud + (size_t)b * NC;
  float ss = 0.f;
  for (int i = lane; i < NC; i += 32) { float v = row[i]; ss += v * v; }
#pragma unroll
  for (int m = 16; m >= 1; m >>= 1) ss += __shfl_xor(ss, m, 32);
  const float scale = 1.0f / fmaxf(sqrtf(ss), 1e-12f);
  _Float16* o = aud16 + (size_t)b * NC;
  for (int i = lane; i < NC; i += 32) o[i] = (_Float16)(row[i] * scale);
}

// --- kernel 2: channel-L2-normalize + transpose -> imgT16[n][hw][c] f16 ---
__global__ void norm_img_kernel(const float* __restrict__ img,
                                _Float16* __restrict__ imgT16) {
  __shared__ float sc[HW];
  __shared__ float tile[32 * HW];   // 24.5 KB
  const int n = blockIdx.x, tid = threadIdx.x;
  const float* base = img + (size_t)n * NC * HW;

  if (tid < HW) {                   // coalesced across threads at fixed c
    float ss = 0.f;
    const float* p = base + tid;
    for (int c = 0; c < NC; ++c) { float v = p[(size_t)c * HW]; ss += v * v; }
    sc[tid] = 1.0f / fmaxf(sqrtf(ss), 1e-12f);
  }
  __syncthreads();

  for (int c0 = 0; c0 < NC; c0 += 32) {
    for (int i = tid; i < 32 * HW; i += 256)       // tile[c*HW+hw] = img[..]
      tile[i] = base[(size_t)c0 * HW + i];
    __syncthreads();
    for (int i = tid; i < HW * 32; i += 256) {     // contiguous f16 writes
      int hw = i >> 5, c = i & 31;
      imgT16[((size_t)n * HW + hw) * NC + c0 + c] =
          (_Float16)(tile[c * HW + hw] * sc[hw]);
    }
    __syncthreads();
  }
}

// --- kernel 3: fused WMMA GEMM + reductions -------------------------------
// grid (4, 256): blockIdx.x = m-tile of 64 cols, blockIdx.y = n
// 128 threads = 4 waves; wave w owns columns m0 = mt*64 + w*16 .. +15
__global__ void __launch_bounds__(128)
av_corr_kernel(const _Float16* __restrict__ imgT,
               const _Float16* __restrict__ aud16,
               float* __restrict__ out) {
  __shared__ _Float16 As[2][HWP * RS];   // 2 x 16.6 KB double-buffered A-tile

  const int n    = blockIdx.y;
  const int mt   = blockIdx.x;
  const int tid  = threadIdx.x;
  const int lane = tid & 31;
  const int half = lane >> 4;
  const int lr   = lane & 15;
  const int wave = tid >> 5;
  const int m0   = mt * 64 + wave * 16;

  float* oA   = out + OFF_A;
  float* oLog = out + OFF_LOG;
  float* oPos = out + OFF_POS;
  float* oNeg = out + OFF_NEG;
  float* oRef = out + OFF_REF;

  v8f acc[NT];
  v8f zero = {};
#pragma unroll
  for (int t = 0; t < NT; ++t) acc[t] = zero;

  const _Float16* imgN = imgT + (size_t)n * HW * NC;
  // B fragment source: lane holds column m0+lr, K rows (kb + half*8 .. +7, kb + 16 + half*8 .. +7)
  const _Float16* audC = aud16 + (size_t)(m0 + lr) * NC + half * 8;

  // zero the 12 padded rows (196..207) of both buffers once
  for (int i = tid; i < 2 * (HWP - HW) * RS; i += 128) {
    int b = i / ((HWP - HW) * RS);
    int r = i - b * ((HWP - HW) * RS);
    As[b][HW * RS + r] = (_Float16)0.f;
  }

  // prologue: stage K-slice 0 into buffer 0, preload B frag 0
  stage_async(&As[0][0], imgN, tid);
  v8h b_lo = *(const v8h*)(audC);
  v8h b_hi = *(const v8h*)(audC + 16);
  wait_async0();
  __syncthreads();

  for (int ks = 0; ks < NC / 32; ++ks) {
    const _Float16* cur = &As[ks & 1][0];

    // prefetch next K-step: B frag (global, pipelined) + A tile (async->LDS)
    v8h nb_lo = {}, nb_hi = {};
    if (ks + 1 < NC / 32) {
      nb_lo = *(const v8h*)(audC + (ks + 1) * 32);
      nb_hi = *(const v8h*)(audC + (ks + 1) * 32 + 16);
      stage_async(&As[(ks + 1) & 1][0], imgN + (ks + 1) * 32, tid);
    }

    v16h bf = cat16(b_lo, b_hi);

    // software-pipelined A-fragment loads from LDS
    // lane layout: row = lr; lanes 0-15: K {0..7,16..23}; 16-31: {8..15,24..31}
    const _Float16* ar0 = cur + (size_t)lr * RS + half * 8;
    v8h a_lo = *(const v8h*)(ar0);
    v8h a_hi = *(const v8h*)(ar0 + 16);
#pragma unroll
    for (int t = 0; t < NT; ++t) {
      v8h p_lo = {}, p_hi = {};
      if (t + 1 < NT) {
        const _Float16* arn = cur + (size_t)((t + 1) * 16 + lr) * RS + half * 8;
        p_lo = *(const v8h*)(arn);
        p_hi = *(const v8h*)(arn + 16);
      }
      v16h af = cat16(a_lo, a_hi);
      acc[t] = __builtin_amdgcn_wmma_f32_16x16x32_f16(
          false, af, false, bf, (short)0, acc[t], false, false);
      a_lo = p_lo; a_hi = p_hi;
    }

    b_lo = nb_lo; b_hi = nb_hi;
    wait_async0();      // next buffer fully landed in LDS
    __syncthreads();    // all waves done reading cur / staging next
  }

  // --- epilogue: per-column reductions over hw ---
  // D layout: acc[t][r] = S[hw = t*16 + half*8 + r, m = m0 + lr]
  float cmax = -1e30f, num = 0.f, den = 0.f;
#pragma unroll
  for (int t = 0; t < NT; ++t) {
#pragma unroll
    for (int r = 0; r < 8; ++r) {
      int hw = t * 16 + half * 8 + r;
      if (hw < HW) {
        float s = acc[t][r];
        float p = sigf((s - EPS_) / TAU_);
        num += p * s; den += p;
        cmax = fmaxf(cmax, s);
      }
    }
  }
  num += __shfl_xor(num, 16, 32);
  den += __shfl_xor(den, 16, 32);
  cmax = fmaxf(cmax, __shfl_xor(cmax, 16, 32));

  if (half == 0) {
    const int m = m0 + lr;
    float simv = (num / den) * ((m == n) ? -99.0f : 1.0f);
    oLog[(size_t)n * (NB + 2) + 1 + m] = simv / LOGIT_T_;
    oRef[(size_t)n * NB + m] = cmax;
  }

  // --- diagonal column m == n: emit A/Pos/Neg maps + sim1/sim2 ---
  if (m0 <= n && n < m0 + 16) {
    const int c = n - m0;
    float sPA = 0.f, sP = 0.f, sNA = 0.f, sN = 0.f;
    if (lr == c) {
#pragma unroll
      for (int t = 0; t < NT; ++t) {
#pragma unroll
        for (int r = 0; r < 8; ++r) {
          int hw = t * 16 + half * 8 + r;
          if (hw < HW) {
            float s    = acc[t][r];
            float pos  = sigf((s - EPS_) / TAU_);
            float pos2 = sigf((s - EPS2_) / TAU_);
            float neg  = 1.0f - pos2;
            oA[(size_t)n * HW + hw]   = s;
            oPos[(size_t)n * HW + hw] = pos;
            oNeg[(size_t)n * HW + hw] = neg;
            sPA += pos * s; sP += pos;
            sNA += neg * s; sN += neg;
          }
        }
      }
    }
    sPA += __shfl_xor(sPA, 16, 32);
    sP  += __shfl_xor(sP , 16, 32);
    sNA += __shfl_xor(sNA, 16, 32);
    sN  += __shfl_xor(sN , 16, 32);
    if (lr == c && half == 0) {
      oLog[(size_t)n * (NB + 2) + 0]        = (sPA / sP) / LOGIT_T_;
      oLog[(size_t)n * (NB + 2) + (NB + 1)] = (sNA / sN) / LOGIT_T_;
    }
  }
}

extern "C" void kernel_launch(void* const* d_in, const int* in_sizes, int n_in,
                              void* d_out, int out_size, void* d_ws, size_t ws_size,
                              hipStream_t stream) {
  const float* img = (const float*)d_in[0];   // [256,512,14,14]
  const float* aud = (const float*)d_in[1];   // [256,512]
  float* out = (float*)d_out;                 // 282112 floats

  // workspace: aud16 (256KB) | imgT16 (49 MB)
  _Float16* aud16  = (_Float16*)d_ws;
  _Float16* imgT16 = (_Float16*)((char*)d_ws + 262144);

  norm_aud_kernel<<<256, 32, 0, stream>>>(aud, aud16);
  norm_img_kernel<<<256, 256, 0, stream>>>(img, imgT16);
  av_corr_kernel<<<dim3(4, 256), 128, 0, stream>>>(imgT16, aud16, out);
}
